// Rwkv7SelfAttention_53429393162446
// MI455X (gfx1250) — matrix operand order repaired and checked by reading, not compiled, not run on hardware
//
#include <hip/hip_runtime.h>
#include <math.h>
#include <stdint.h>

// ---------------------------------------------------------------------------
// CDNA5 async copy support (guarded: falls back to sync LDS staging if the
// toolchain does not declare the async builtins).
// ---------------------------------------------------------------------------
#if defined(__AMDGCN__) && __has_builtin(__builtin_amdgcn_global_load_async_to_lds_b128)
#define HAVE_ASYNC_LDS 1
#else
#define HAVE_ASYNC_LDS 0
#endif

#if HAVE_ASYNC_LDS
#if __has_builtin(__builtin_amdgcn_s_wait_asynccnt)
#define WAIT_ASYNC(n) __builtin_amdgcn_s_wait_asynccnt(n)
#else
#define WAIT_ASYNC(n) asm volatile("s_wait_asynccnt " #n ::: "memory")
#endif
#else
#define WAIT_ASYNC(n)
#endif

namespace {

constexpr int kH  = 2048;
constexpr int kHS = 64;
constexpr int kNH = 32;
constexpr int kT  = 1024;
constexpr int kDD = 64;
constexpr int kDA = 64;
constexpr int kDG = 128;
constexpr int kDV = 32;

typedef __attribute__((ext_vector_type(16))) _Float16 v16h;
typedef __attribute__((ext_vector_type(8)))  float    v8f;
typedef __attribute__((ext_vector_type(4)))  int      v4i_;

__device__ __forceinline__ float sigm(float x) { return 1.0f / (1.0f + expf(-x)); }

#if HAVE_ASYNC_LDS
__device__ __forceinline__ void async_cp16(const _Float16* g, _Float16* lds) {
  // Builtin signature (from toolchain diagnostic): param0 is int4 in AS1;
  // LDS side is AS3 (32-bit pointers; generic LDS addr low bits = LDS offset).
  __builtin_amdgcn_global_load_async_to_lds_b128(
      (__attribute__((address_space(1))) v4i_*)(uintptr_t)g,
      (__attribute__((address_space(3))) v4i_*)(uint32_t)(uintptr_t)lds,
      0, 0);
}
#endif

// ---------------------------------------------------------------------------
// f32 -> f16 conversion, 4 elements per thread.
// ---------------------------------------------------------------------------
__global__ __launch_bounds__(256)
void cvt_f16_kernel(const float* __restrict__ src, _Float16* __restrict__ dst) {
  const size_t i = ((size_t)blockIdx.x * 256 + threadIdx.x) * 4;
  const float4 v = *(const float4*)(src + i);
  union { _Float16 h[4]; uint2 u; } o;
  o.h[0] = (_Float16)v.x; o.h[1] = (_Float16)v.y;
  o.h[2] = (_Float16)v.z; o.h[3] = (_Float16)v.w;
  *(uint2*)(dst + i) = o.u;
}

// ---------------------------------------------------------------------------
// LayerNorm over H per timestep. One block (256 thr = 8 waves) per row.
// ---------------------------------------------------------------------------
__global__ __launch_bounds__(256)
void ln_kernel(const float* __restrict__ x, const float* __restrict__ w,
               const float* __restrict__ b, float* __restrict__ y, float eps) {
  __shared__ float red[8];
  const int t = blockIdx.x;
  const float* row = x + (size_t)t * kH;
  float s = 0.f;
  for (int j = threadIdx.x; j < kH; j += 256) s += row[j];
  for (int o = 16; o; o >>= 1) s += __shfl_xor(s, o, 32);
  if ((threadIdx.x & 31) == 0) red[threadIdx.x >> 5] = s;
  __syncthreads();
  float mean = 0.f;
  for (int i = 0; i < 8; ++i) mean += red[i];
  mean *= (1.0f / kH);
  __syncthreads();
  float v = 0.f;
  for (int j = threadIdx.x; j < kH; j += 256) { float d = row[j] - mean; v += d * d; }
  for (int o = 16; o; o >>= 1) v += __shfl_xor(v, o, 32);
  if ((threadIdx.x & 31) == 0) red[threadIdx.x >> 5] = v;
  __syncthreads();
  float var = 0.f;
  for (int i = 0; i < 8; ++i) var += red[i];
  var *= (1.0f / kH);
  const float inv = rsqrtf(var + eps);
  float* yr = y + (size_t)t * kH;
  for (int j = threadIdx.x; j < kH; j += 256)
    yr[j] = (row[j] - mean) * inv * w[j] + b[j];
}

// ---------------------------------------------------------------------------
// Time-shift mixing. xr/xk are consumed only by WMMA GEMMs -> emit f16 only.
// xv feeds both the GEMM (f16) and the v-gate MLP (f32). xw/xa/xg stay f32.
// ---------------------------------------------------------------------------
__global__ __launch_bounds__(256)
void mix_kernel(const float* __restrict__ xn, const float* __restrict__ state1,
                const float* __restrict__ mr, const float* __restrict__ mw,
                const float* __restrict__ mk, const float* __restrict__ mv,
                const float* __restrict__ ma, const float* __restrict__ mg,
                _Float16* __restrict__ xr16, float* __restrict__ xw,
                _Float16* __restrict__ xk16, float* __restrict__ xv,
                _Float16* __restrict__ xv16, float* __restrict__ xa,
                float* __restrict__ xg, float* __restrict__ s1out) {
  const int idx = blockIdx.x * 256 + threadIdx.x;
  const int t = idx / kH, j = idx % kH;
  const float cur  = xn[idx];
  const float past = (t == 0) ? state1[j] : xn[idx - kH];
  const float sx = past - cur;
  xr16[idx] = (_Float16)(cur + mr[j] * sx);
  xw[idx]   = cur + mw[j] * sx;
  xk16[idx] = (_Float16)(cur + mk[j] * sx);
  const float vv = cur + mv[j] * sx;
  xv[idx]   = vv;
  xv16[idx] = (_Float16)vv;
  xa[idx]   = cur + ma[j] * sx;
  xg[idx]   = cur + mg[j] * sx;
  if (t == kT - 1) s1out[j] = cur;
}

// ---------------------------------------------------------------------------
// C[M,N] = A[M,K] @ W[N,K]^T (+ optional residual). A, W pre-converted f16.
// Block: 256 thr = 8 waves; tile 64(M) x 128(N); wave = 16x64 (4 WMMA accs).
// Double-buffered LDS stages filled with GLOBAL_LOAD_ASYNC_TO_LDS_B128
// (ASYNCcnt) when available, else synchronous 16B copies.
// ---------------------------------------------------------------------------
__global__ __launch_bounds__(256)
void gemm_wmma(const _Float16* __restrict__ A, const _Float16* __restrict__ W,
               const float* __restrict__ resid, float* __restrict__ C,
               int M, int N, int K) {
  __shared__ _Float16 As[2][64][40];    // [m][k], 80B row stride (16B aligned)
  __shared__ _Float16 Bs[2][128][40];   // [n][k]
  const int m0 = blockIdx.y * 64;
  const int n0 = blockIdx.x * 128;
  const int tid  = threadIdx.x;
  const int wave = tid >> 5;
  const int lane = tid & 31;
  const int wm   = wave >> 1;           // 0..3 : M subtile
  const int wn   = wave & 1;            // 0..1 : 64-wide N half
  const int l16  = lane & 15;
  const int kg   = lane >> 4;           // K-half select (contiguous: k = 16*kg + s)

  // Copy geometry: 16B chunks. A: 256 chunks (1/thread). B: 512 (2/thread).
  const int cr = tid >> 2;              // 0..63
  const int cc = (tid & 3) * 8;         // halves

  v8f acc0 = {}, acc1 = {}, acc2 = {}, acc3 = {};

  auto stage = [&](int buf, int k0) {
    const _Float16* ga  = A + (size_t)(m0 + cr) * K + k0 + cc;
    const _Float16* gb0 = W + (size_t)(n0 + cr) * K + k0 + cc;
    const _Float16* gb1 = W + (size_t)(n0 + 64 + cr) * K + k0 + cc;
#if HAVE_ASYNC_LDS
    async_cp16(ga,  &As[buf][cr][cc]);
    async_cp16(gb0, &Bs[buf][cr][cc]);
    async_cp16(gb1, &Bs[buf][64 + cr][cc]);
#else
    *(float4*)&As[buf][cr][cc]      = *(const float4*)ga;
    *(float4*)&Bs[buf][cr][cc]      = *(const float4*)gb0;
    *(float4*)&Bs[buf][64 + cr][cc] = *(const float4*)gb1;
#endif
  };

  const int nk = K >> 5;
  stage(0, 0);
  for (int kk = 0; kk < nk; ++kk) {
    if (kk + 1 < nk) {
      stage((kk + 1) & 1, (kk + 1) * 32);
      WAIT_ASYNC(3);   // retire stage kk (3 async ops/stage just issued)
    } else {
      WAIT_ASYNC(0);
    }
    __syncthreads();

    const int buf = kk & 1;
    // A fragment: 16x32, lane l16 = row, kg selects K sub-block.
    v16h af;
    #pragma unroll
    for (int p = 0; p < 8; ++p) {
      const int koff = 2 * p + 16 * kg;
      af[2 * p]     = As[buf][wm * 16 + l16][koff];
      af[2 * p + 1] = As[buf][wm * 16 + l16][koff + 1];
    }
    // B fragments: 32x16 each, lane l16 = column, kg selects K half.
    v16h bf0, bf1, bf2, bf3;
    #pragma unroll
    for (int p = 0; p < 8; ++p) {
      const int kb = 2 * p + 16 * kg;
      const int nb = wn * 64 + l16;
      bf0[2 * p] = Bs[buf][nb][kb];      bf0[2 * p + 1] = Bs[buf][nb][kb + 1];
      bf1[2 * p] = Bs[buf][nb + 16][kb]; bf1[2 * p + 1] = Bs[buf][nb + 16][kb + 1];
      bf2[2 * p] = Bs[buf][nb + 32][kb]; bf2[2 * p + 1] = Bs[buf][nb + 32][kb + 1];
      bf3[2 * p] = Bs[buf][nb + 48][kb]; bf3[2 * p + 1] = Bs[buf][nb + 48][kb + 1];
    }
    acc0 = __builtin_amdgcn_wmma_f32_16x16x32_f16(false, af, false, bf0,
                                                  (short)0, acc0, false, false);
    acc1 = __builtin_amdgcn_wmma_f32_16x16x32_f16(false, af, false, bf1,
                                                  (short)0, acc1, false, false);
    acc2 = __builtin_amdgcn_wmma_f32_16x16x32_f16(false, af, false, bf2,
                                                  (short)0, acc2, false, false);
    acc3 = __builtin_amdgcn_wmma_f32_16x16x32_f16(false, af, false, bf3,
                                                  (short)0, acc3, false, false);
    __syncthreads();
  }

  // D layout: VGPR q -> row (q + 8*kg), col l16.
  #pragma unroll
  for (int q = 0; q < 8; ++q) {
    const int row = m0 + wm * 16 + q + 8 * kg;
    const int c0  = n0 + wn * 64 + l16;
    float o0 = acc0[q], o1 = acc1[q], o2 = acc2[q], o3 = acc3[q];
    if (resid != nullptr) {
      o0 += resid[(size_t)row * N + c0];
      o1 += resid[(size_t)row * N + c0 + 16];
      o2 += resid[(size_t)row * N + c0 + 32];
      o3 += resid[(size_t)row * N + c0 + 48];
    }
    C[(size_t)row * N + c0]      = o0;
    C[(size_t)row * N + c0 + 16] = o1;
    C[(size_t)row * N + c0 + 32] = o2;
    C[(size_t)row * N + c0 + 48] = o3;
  }
}

// ---------------------------------------------------------------------------
// Stage-1 MLP: h[t,d] = act( sum_j A[t,j] * Wm[j,d] ), Wm is (H, D).
// act: 0=none, 1=tanh, 2=sigmoid
// ---------------------------------------------------------------------------
__global__ __launch_bounds__(256)
void mlp1_kernel(const float* __restrict__ A, const float* __restrict__ Wm,
                 float* __restrict__ h, int D, int act) {
  const int idx = blockIdx.x * 256 + threadIdx.x;    // T*D
  const int t = idx / D, d = idx % D;
  const float* row = A + (size_t)t * kH;
  float s = 0.f;
  for (int j = 0; j < kH; ++j) s += row[j] * Wm[(size_t)j * D + d];
  if (act == 1) s = tanhf(s);
  else if (act == 2) s = sigm(s);
  h[idx] = s;
}

// ---------------------------------------------------------------------------
// Stage-2 MLP: s = sum_d h[t,d] * Wm[d,j]; epilogue per mode.
// mode 0: out = s                                  (g gate)
// mode 1: out = exp(-0.606531*sigmoid(bias[j]+s))  (decay w)
// mode 2: out = sigmoid(bias[j]+s)                 (a)
// mode 3: p = sigmoid(bias[j]+s); out = v + (v_first - v)*p
// ---------------------------------------------------------------------------
__global__ __launch_bounds__(256)
void mlp2_kernel(const float* __restrict__ h, const float* __restrict__ Wm,
                 const float* __restrict__ bias, const float* __restrict__ v_in,
                 const float* __restrict__ v_first, float* __restrict__ out,
                 int D, int mode) {
  const int idx = blockIdx.x * 256 + threadIdx.x;    // T*H
  const int t = idx / kH, j = idx % kH;
  const float* hr = h + (size_t)t * D;
  float s = 0.f;
  for (int d = 0; d < D; ++d) s += hr[d] * Wm[(size_t)d * kH + j];
  float o;
  if (mode == 0)      o = s;
  else if (mode == 1) o = expf(-0.606531f * sigm(bias[j] + s));
  else if (mode == 2) o = sigm(bias[j] + s);
  else {
    const float p = sigm(bias[j] + s);
    const float v = v_in[idx];
    o = v + (v_first[idx] - v) * p;
  }
  out[idx] = o;
}

// ---------------------------------------------------------------------------
// Per-(t,head): kk = normalize(k*k_k); b = kk*a; aseq = -kk;
// kmod = k*(1+(a-1)*k_a) written in place over k. 64 threads per (t,head).
// ---------------------------------------------------------------------------
__global__ __launch_bounds__(64)
void coeff_kernel(float* __restrict__ k, const float* __restrict__ a,
                  const float* __restrict__ k_k, const float* __restrict__ k_a,
                  float* __restrict__ aseq, float* __restrict__ bseq) {
  __shared__ float csh[2];
  const int th = blockIdx.x;          // t*NH + h
  const int hh = th % kNH;
  const int i  = threadIdx.x;
  const int base = th * kHS;
  const float kv  = k[base + i];
  float kkv = kv * k_k[hh * kHS + i];
  float ss = kkv * kkv;
  for (int o = 16; o; o >>= 1) ss += __shfl_xor(ss, o, 32);
  if ((i & 31) == 0) csh[i >> 5] = ss;
  __syncthreads();
  const float norm = sqrtf(csh[0] + csh[1]) + 1e-12f;
  kkv /= norm;
  const float av = a[base + i];
  bseq[base + i] = kkv * av;
  aseq[base + i] = -kkv;
  k[base + i]    = kv * (1.0f + (av - 1.0f) * k_a[hh * kHS + i]);
}

// ---------------------------------------------------------------------------
// Sequential WKV scan. One block (64 thr = 2 waves) per head; thread i owns
// row i of the 64x64 state in registers. Fused head group-norm, ln_x affine,
// bonus term and g-gate; writes yg in f16 (input of the output projection).
// ---------------------------------------------------------------------------
__global__ __launch_bounds__(64)
void scan_kernel(const float* __restrict__ state2, const float* __restrict__ wdec,
                 const float* __restrict__ kmod, const float* __restrict__ v,
                 const float* __restrict__ aseq, const float* __restrict__ bseq,
                 const float* __restrict__ r, const float* __restrict__ r_k,
                 const float* __restrict__ ln_x_w, const float* __restrict__ ln_x_b,
                 const float* __restrict__ g, _Float16* __restrict__ yg,
                 float* __restrict__ state2_out) {
  const int hh = blockIdx.x;
  const int i  = threadIdx.x;
  __shared__ float sw[kHS], sk[kHS], sv[kHS], sa[kHS], sb[kHS], sr[kHS];
  __shared__ float red[6];

  float S[kHS];
  #pragma unroll
  for (int j = 0; j < kHS; ++j)
    S[j] = state2[((size_t)hh * kHS + i) * kHS + j];

  const float rkv = r_k[hh * kHS + i];
  const float lnw = ln_x_w[hh * kHS + i];
  const float lnb = ln_x_b[hh * kHS + i];
  const int wid = i >> 5;

  for (int t = 0; t < kT; ++t) {
    const int base = (t * kNH + hh) * kHS;
    sw[i] = wdec[base + i];
    sk[i] = kmod[base + i];
    sv[i] = v[base + i];
    sa[i] = aseq[base + i];
    sb[i] = bseq[base + i];
    sr[i] = r[base + i];
    __syncthreads();

    float sav = 0.f;
    #pragma unroll
    for (int j = 0; j < kHS; ++j) {
      S[j] *= sw[j];
      sav += S[j] * sa[j];
    }
    const float vi = sv[i];
    float outv = 0.f;
    #pragma unroll
    for (int j = 0; j < kHS; ++j) {
      S[j] += sav * sb[j] + vi * sk[j];
      outv += S[j] * sr[j];
    }

    // Fused reductions: sum(out), sum(out^2), bonus dot(r*k*r_k).
    float p0 = outv, p1 = outv * outv, p2 = sr[i] * sk[i] * rkv;
    for (int o = 16; o; o >>= 1) {
      p0 += __shfl_xor(p0, o, 32);
      p1 += __shfl_xor(p1, o, 32);
      p2 += __shfl_xor(p2, o, 32);
    }
    if ((i & 31) == 0) { red[wid * 3] = p0; red[wid * 3 + 1] = p1; red[wid * 3 + 2] = p2; }
    __syncthreads();
    const float sum   = red[0] + red[3];
    const float sumsq = red[1] + red[4];
    const float dot   = red[2] + red[5];
    const float mean = sum * (1.0f / kHS);
    const float var  = sumsq * (1.0f / kHS) - mean * mean;
    const float yn   = (outv - mean) * rsqrtf(var + 0.00064f);
    const float yv = (yn * lnw + lnb + dot * vi) * g[(size_t)t * kH + hh * kHS + i];
    yg[(size_t)t * kH + hh * kHS + i] = (_Float16)yv;
  }

  #pragma unroll
  for (int j = 0; j < kHS; ++j)
    state2_out[((size_t)hh * kHS + i) * kHS + j] = S[j];
}

}  // namespace

// ---------------------------------------------------------------------------
extern "C" void kernel_launch(void* const* d_in, const int* in_sizes, int n_in,
                              void* d_out, int out_size, void* d_ws, size_t ws_size,
                              hipStream_t stream) {
  (void)in_sizes; (void)n_in; (void)out_size; (void)ws_size;

  const float* x       = (const float*)d_in[0];
  const float* state1  = (const float*)d_in[1];
  const float* state2  = (const float*)d_in[2];
  const float* v_first = (const float*)d_in[3];
  const float* x_r = (const float*)d_in[4];
  const float* x_w = (const float*)d_in[5];
  const float* x_k = (const float*)d_in[6];
  const float* x_v = (const float*)d_in[7];
  const float* x_a = (const float*)d_in[8];
  const float* x_g = (const float*)d_in[9];
  const float* W_r = (const float*)d_in[10];
  const float* W_k = (const float*)d_in[11];
  const float* W_v = (const float*)d_in[12];
  const float* W_o = (const float*)d_in[13];
  const float* w0  = (const float*)d_in[14];
  const float* w1  = (const float*)d_in[15];
  const float* w2  = (const float*)d_in[16];
  const float* a0  = (const float*)d_in[17];
  const float* a1  = (const float*)d_in[18];
  const float* a2  = (const float*)d_in[19];
  const float* v0  = (const float*)d_in[20];
  const float* v1  = (const float*)d_in[21];
  const float* v2  = (const float*)d_in[22];
  const float* g1  = (const float*)d_in[23];
  const float* g2  = (const float*)d_in[24];
  const float* k_k = (const float*)d_in[25];
  const float* k_a = (const float*)d_in[26];
  const float* r_k = (const float*)d_in[27];
  const float* ln_x_w = (const float*)d_in[28];
  const float* ln_x_b = (const float*)d_in[29];
  const float* ln1_w  = (const float*)d_in[30];
  const float* ln1_b  = (const float*)d_in[31];

  const size_t TH = (size_t)kT * kH;
  const size_t HH = (size_t)kH * kH;

  // -------- workspace carve --------
  // f32 buffers (8 x T*H) + MLP hidden states, then f16 region.
  float* ws   = (float*)d_ws;
  float* xn   = ws + 0 * TH;
  float* xw   = ws + 1 * TH;
  float* xv   = ws + 2 * TH;
  float* xa   = ws + 3 * TH;
  float* xg   = ws + 4 * TH;
  float* rbuf = ws + 5 * TH;
  float* kbuf = ws + 6 * TH;   // k, overwritten in place with kmod
  float* vbuf = ws + 7 * TH;
  float* h_w  = ws + 8 * TH;                 // T*64
  float* h_a  = h_w + (size_t)kT * kDD;      // T*64
  float* h_v  = h_a + (size_t)kT * kDA;      // T*32
  float* h_g  = h_v + (size_t)kT * kDV;      // T*128
  _Float16* f16 = (_Float16*)(h_g + (size_t)kT * kDG);
  _Float16* Wr16 = f16;
  _Float16* Wk16 = Wr16 + HH;
  _Float16* Wv16 = Wk16 + HH;
  _Float16* Wo16 = Wv16 + HH;
  _Float16* xr16 = Wo16 + HH;
  _Float16* xk16 = xr16 + TH;
  _Float16* xv16 = xk16 + TH;
  _Float16* yg16 = xv16 + TH;
  // Aliases for the post-GEMM phase (those f32 buffers are dead by then):
  float* wdec = xn;
  float* abuf = xa;
  float* gbuf = xg;
  float* aseq = xw;
  float* bseq = xv;

  float* out_y  = (float*)d_out;                    // T*H
  float* out_s1 = out_y + TH;                       // H
  float* out_s2 = out_s1 + kH;                      // NH*HS*HS
  float* out_vf = out_s2 + (size_t)kNH * kHS * kHS; // T*H

  // 1. Weight conversion to f16 (once per launch; halves tile read traffic).
  const int cvtW = (int)(HH / 1024);
  cvt_f16_kernel<<<cvtW, 256, 0, stream>>>(W_r, Wr16);
  cvt_f16_kernel<<<cvtW, 256, 0, stream>>>(W_k, Wk16);
  cvt_f16_kernel<<<cvtW, 256, 0, stream>>>(W_v, Wv16);
  cvt_f16_kernel<<<cvtW, 256, 0, stream>>>(W_o, Wo16);

  // 2. LayerNorm (ln1) into xn.
  ln_kernel<<<kT, 256, 0, stream>>>(x, ln1_w, ln1_b, xn, 1e-5f);

  // 3. Time-shift mixes + state1_out (f16 A-matrices emitted directly).
  mix_kernel<<<(int)(TH / 256), 256, 0, stream>>>(
      xn, state1, x_r, x_w, x_k, x_v, x_a, x_g,
      xr16, xw, xk16, xv, xv16, xa, xg, out_s1);

  // 4. Big projections via WMMA (async double-buffered LDS): r, k, v.
  dim3 ggrid(kH / 128, kT / 64);
  gemm_wmma<<<ggrid, 256, 0, stream>>>(xr16, Wr16, nullptr, rbuf, kT, kH, kH);
  gemm_wmma<<<ggrid, 256, 0, stream>>>(xk16, Wk16, nullptr, kbuf, kT, kH, kH);
  gemm_wmma<<<ggrid, 256, 0, stream>>>(xv16, Wv16, nullptr, vbuf, kT, kH, kH);

  // 5. Gate MLPs, stage 1 (down-projections).
  mlp1_kernel<<<kT * kDD / 256, 256, 0, stream>>>(xw, w1, h_w, kDD, 1); // tanh
  mlp1_kernel<<<kT * kDA / 256, 256, 0, stream>>>(xa, a1, h_a, kDA, 0);
  mlp1_kernel<<<kT * kDV / 256, 256, 0, stream>>>(xv, v1, h_v, kDV, 0);
  mlp1_kernel<<<kT * kDG / 256, 256, 0, stream>>>(xg, g1, h_g, kDG, 2); // sigmoid

  // 6. Gate MLPs, stage 2 (up-projections + fused epilogues).
  const int g2blocks = (int)(TH / 256);
  mlp2_kernel<<<g2blocks, 256, 0, stream>>>(h_g, g2, nullptr, nullptr, nullptr,
                                            gbuf, kDG, 0);
  mlp2_kernel<<<g2blocks, 256, 0, stream>>>(h_w, w2, w0, nullptr, nullptr,
                                            wdec, kDD, 1);
  mlp2_kernel<<<g2blocks, 256, 0, stream>>>(h_a, a2, a0, nullptr, nullptr,
                                            abuf, kDA, 2);
  mlp2_kernel<<<g2blocks, 256, 0, stream>>>(h_v, v2, v0, vbuf, v_first,
                                            vbuf, kDV, 3);

  // 7. Per-(t,head) scan coefficients (kmod in place over kbuf).
  coeff_kernel<<<kT * kNH, 64, 0, stream>>>(kbuf, abuf, k_k, k_a, aseq, bseq);

  // 8. Sequential WKV scan + fused group-norm/bonus/gate; writes state2_out.
  scan_kernel<<<kNH, 64, 0, stream>>>(state2, wdec, kbuf, vbuf, aseq, bseq,
                                      rbuf, r_k, ln_x_w, ln_x_b, gbuf,
                                      yg16, out_s2);

  // 9. Output projection with fused residual: out = x + yg @ W_o^T.
  gemm_wmma<<<ggrid, 256, 0, stream>>>(yg16, Wo16, x, out_y, kT, kH, kH);

  // 10. v_first passthrough output.
  (void)hipMemcpyAsync(out_vf, v_first, TH * sizeof(float),
                       hipMemcpyDeviceToDevice, stream);
}
